// CausalSelfAttention_65008624992421
// MI455X (gfx1250) — compile-verified, matrix operand order checked by reading, not run on hardware
//
#include <hip/hip_runtime.h>
#include <hip/hip_bf16.h>

#define BATCH 2
#define SEQ   2048
#define EMB   1024
#define HEADS 16
#define HD    64
#define E3    3072

typedef __attribute__((ext_vector_type(16))) __bf16 v16bf;
typedef __attribute__((ext_vector_type(8)))  __bf16 v8bf;
typedef __attribute__((ext_vector_type(4)))  __bf16 v4bf;
typedef __attribute__((ext_vector_type(8)))  float  v8f;
typedef __attribute__((ext_vector_type(4)))  float  v4f;

// Fragment load for A (tile [M][K]) and B (tile [N][K]): two ds_load_b128.
__device__ __forceinline__ v16bf load_frag(const __bf16* base, int idx0, int ld) {
  const int lane = threadIdx.x & 31;
  const __bf16* p = base + (size_t)(idx0 + (lane & 15)) * ld + (lane >> 4) * 8;
  v8bf lo = *(const v8bf*)p;          // K = kh*8 + 0..7
  v8bf hi = *(const v8bf*)(p + 16);   // K = 16 + kh*8 + 0..7
  return __builtin_shufflevector(lo, hi, 0, 1, 2, 3, 4, 5, 6, 7,
                                 8, 9, 10, 11, 12, 13, 14, 15);
}

__device__ __forceinline__ v8f wmma_bf16(v16bf a, v16bf b, v8f c) {
  return __builtin_amdgcn_wmma_f32_16x16x32_bf16(false, a, false, b, (short)0, c,
                                                 false, false);
}

// ---- Async memory->LDS copy (gfx1250 GLOBAL_LOAD_ASYNC_TO_LDS, ASYNCcnt) ---
__device__ __forceinline__ void async_b128(uint32_t lds_off, const __bf16* gsrc) {
  asm volatile("global_load_async_to_lds_b128 %0, %1, off"
               :: "v"(lds_off), "v"(gsrc));
}
__device__ __forceinline__ void wait_async() {
  asm volatile("s_wait_asynccnt 0" ::: "memory");
}

// ROWS x (CHUNKS*8) bf16 tile: async global->LDS copy, 16B per op.
template <int ROWS, int CHUNKS>
__device__ __forceinline__ void stage_async_bf16(__bf16* dst, const __bf16* src,
                                                 size_t src_ld, int ldd) {
  const int t = threadIdx.x;
  const uint32_t base = (uint32_t)(uintptr_t)dst;  // LDS aperture: low 32 = offset
#pragma unroll
  for (int i = t; i < ROWS * CHUNKS; i += 128) {
    int r = i / CHUNKS, c8 = (i % CHUNKS) * 8;
    async_b128(base + (uint32_t)(r * ldd + c8) * 2u, src + (size_t)r * src_ld + c8);
  }
}

// ---- Synchronous converting stagers (fp32 source) --------------------------

// ROWS x 32(K) fp32 -> row-major bf16 tile.
template <int ROWS>
__device__ __forceinline__ void stage_a_f32(__bf16* dst, const float* src,
                                            size_t src_ld, int ldd) {
  const int t = threadIdx.x;
#pragma unroll
  for (int i = t; i < ROWS * 8; i += 128) {    // 8 x float4 per row
    int r = i >> 3, c4 = (i & 7) * 4;
    v4f v = *(const v4f*)(src + (size_t)r * src_ld + c4);
    v4bf b = {(__bf16)v[0], (__bf16)v[1], (__bf16)v[2], (__bf16)v[3]};
    *(v4bf*)(dst + r * ldd + c4) = b;
  }
}

// 32(K) x 64(N) fp32 -> LDS [N][K] bf16 via per-thread 4x4 transpose.
__device__ __forceinline__ void stage_bT_f32(__bf16* dst, const float* src,
                                             size_t src_ld, int ldd) {
  const int t = threadIdx.x;
  const int nb = (t & 15) * 4;
  const int kb = (t >> 4) * 4;
  v4f rows[4];
#pragma unroll
  for (int j = 0; j < 4; ++j)
    rows[j] = *(const v4f*)(src + (size_t)(kb + j) * src_ld + nb);
#pragma unroll
  for (int i = 0; i < 4; ++i) {
    v4bf col = {(__bf16)rows[0][i], (__bf16)rows[1][i],
                (__bf16)rows[2][i], (__bf16)rows[3][i]};
    *(v4bf*)(dst + (nb + i) * ldd + kb) = col;
  }
}

// 2x4 WMMA tile: 2 A frags x 4 B frags -> 8 back-to-back WMMAs.
__device__ __forceinline__ void mma_2x4(v8f acc[8], const __bf16* As, int arow,
                                        int lda, const __bf16* Bs, int ldb) {
  v16bf a0 = load_frag(As, arow, lda);
  v16bf a1 = load_frag(As, arow + 16, lda);
  v16bf b0 = load_frag(Bs, 0, ldb);
  v16bf b1 = load_frag(Bs, 16, ldb);
  v16bf b2 = load_frag(Bs, 32, ldb);
  v16bf b3 = load_frag(Bs, 48, ldb);
  acc[0] = wmma_bf16(a0, b0, acc[0]);
  acc[1] = wmma_bf16(a0, b1, acc[1]);
  acc[2] = wmma_bf16(a0, b2, acc[2]);
  acc[3] = wmma_bf16(a0, b3, acc[3]);
  acc[4] = wmma_bf16(a1, b0, acc[4]);
  acc[5] = wmma_bf16(a1, b1, acc[5]);
  acc[6] = wmma_bf16(a1, b2, acc[6]);
  acc[7] = wmma_bf16(a1, b3, acc[7]);
}

// ---------------- Kernel 1: QKV = X @ W_qkv + b
//   Q,K -> [B][H][S][D] bf16 ; V -> transposed [B][H][D][S] bf16
__global__ __launch_bounds__(128) void k_qkv(const float* __restrict__ x,
                                             const float* __restrict__ Wqkv,
                                             const float* __restrict__ bqkv,
                                             __bf16* __restrict__ Q,
                                             __bf16* __restrict__ Kb,
                                             __bf16* __restrict__ Vt) {
  __shared__ __bf16 As[128][40];  // [M][K]
  __shared__ __bf16 Bs[64][40];   // [N][K]
  const int m0 = blockIdx.x * 128;
  const int n0 = blockIdx.y * 64;
  const int tid = threadIdx.x, wave = tid >> 5, lane = tid & 31;
  v8f acc[8] = {};
  for (int k0 = 0; k0 < EMB; k0 += 32) {
    if (k0 + 32 < EMB) {
      __builtin_prefetch(x + (size_t)(m0 + tid) * EMB + k0 + 32, 0, 3);
      __builtin_prefetch(Wqkv + (size_t)(k0 + 32 + (tid >> 2)) * E3 + n0, 0, 3);
    }
    __syncthreads();
    stage_a_f32<128>(&As[0][0], x + (size_t)m0 * EMB + k0, EMB, 40);
    stage_bT_f32(&Bs[0][0], Wqkv + (size_t)k0 * E3 + n0, E3, 40);
    __syncthreads();
    mma_2x4(acc, &As[0][0], wave * 32, 40, &Bs[0][0], 40);
  }
  const int kh = lane >> 4, nn = lane & 15;
  const int which = n0 >> 10;               // SCALAR: whole 64-col tile is Q, K or V
  const int bb    = m0 >> 11;               // SCALAR: block never crosses batch
  const int s_base = (m0 & (SEQ - 1)) + wave * 32 + 8 * kh;
  if (which == 2) {
    // V stored transposed [B][H][D][S]; acc elements are consecutive in s
    // -> one 16B global store per (h2,j).
#pragma unroll
    for (int h2 = 0; h2 < 2; ++h2) {
#pragma unroll
      for (int j = 0; j < 4; ++j) {
        int col = n0 + j * 16 + nn;
        int h = (col >> 6) & 15, d = col & 63;
        float bias = bqkv[col];
        const v8f& c = acc[h2 * 4 + j];
        v8bf pk = {(__bf16)(c[0] + bias), (__bf16)(c[1] + bias),
                   (__bf16)(c[2] + bias), (__bf16)(c[3] + bias),
                   (__bf16)(c[4] + bias), (__bf16)(c[5] + bias),
                   (__bf16)(c[6] + bias), (__bf16)(c[7] + bias)};
        int s = s_base + h2 * 16;
        *(v8bf*)&Vt[((size_t)(bb * HEADS + h) * HD + d) * SEQ + s] = pk;
      }
    }
  } else {
    __bf16* dst = which ? Kb : Q;           // uniform select
#pragma unroll
    for (int h2 = 0; h2 < 2; ++h2) {
#pragma unroll
      for (int j = 0; j < 4; ++j) {
        int col = n0 + j * 16 + nn;
        int h = (col >> 6) & 15, d = col & 63;
        float bias = bqkv[col];
        const v8f& c = acc[h2 * 4 + j];
#pragma unroll
        for (int i = 0; i < 8; ++i) {
          int s = s_base + h2 * 16 + i;
          dst[((size_t)(bb * HEADS + h) * SEQ + s) * HD + d] = (__bf16)(c[i] + bias);
        }
      }
    }
  }
}

// ---------------- Kernel 2: raw masked scores = Q K^T / sqrt(D) -> attnW
__global__ __launch_bounds__(128) void k_scores(const __bf16* __restrict__ Q,
                                                const __bf16* __restrict__ Kb,
                                                float* __restrict__ P) {
  const int bh  = blockIdx.z;
  const int q0  = blockIdx.x * 128;
  const int kk0 = blockIdx.y * 64;
  float* Ph = P + (size_t)bh * SEQ * SEQ;
  const int tid = threadIdx.x, wave = tid >> 5, lane = tid & 31;
  if (kk0 > q0 + 127) {                // fully above diagonal
    for (int i = tid; i < 128 * 16; i += 128) {
      int r = i >> 4, c4 = (i & 15) * 4;
      v4f m = {-1e30f, -1e30f, -1e30f, -1e30f};
      *(v4f*)(Ph + (size_t)(q0 + r) * SEQ + kk0 + c4) = m;
    }
    return;
  }
  __shared__ __bf16 Qs[128][72];       // [q][d]
  __shared__ __bf16 Ks[64][72];        // [k][d] == [N][K]
  stage_async_bf16<128, 8>(&Qs[0][0], Q + ((size_t)bh * SEQ + q0) * HD, HD, 72);
  stage_async_bf16<64, 8>(&Ks[0][0], Kb + ((size_t)bh * SEQ + kk0) * HD, HD, 72);
  wait_async();
  __syncthreads();
  v8f acc[8] = {};
#pragma unroll
  for (int d0 = 0; d0 < HD; d0 += 32)
    mma_2x4(acc, &Qs[0][d0], wave * 32, 72, &Ks[0][d0], 72);
  const int kh = lane >> 4, nn = lane & 15;
  const float scale = 0.125f;          // 1/sqrt(64)
#pragma unroll
  for (int h2 = 0; h2 < 2; ++h2) {
#pragma unroll
    for (int j = 0; j < 4; ++j) {
      const v8f& c = acc[h2 * 4 + j];
#pragma unroll
      for (int i = 0; i < 8; ++i) {
        int q = q0 + wave * 32 + h2 * 16 + i + 8 * kh;
        int k = kk0 + j * 16 + nn;
        Ph[(size_t)q * SEQ + k] = (k <= q) ? c[i] * scale : -1e30f;
      }
    }
  }
}

// ---------------- Kernel 3: row softmax in place (one block per row)
__global__ __launch_bounds__(256) void k_softmax(float* __restrict__ P) {
  __shared__ float buf[SEQ];
  __shared__ float red[9];
  float* row = P + (size_t)blockIdx.x * SEQ;
  const int tid = threadIdx.x;
  float m = -3.0e38f;
  for (int i = tid; i < SEQ; i += 256) {
    float v = row[i];
    buf[i] = v;
    m = fmaxf(m, v);
  }
#pragma unroll
  for (int off = 16; off; off >>= 1) m = fmaxf(m, __shfl_xor(m, off, 32));
  if ((tid & 31) == 0) red[tid >> 5] = m;
  __syncthreads();
  if (tid == 0) {
    float v = red[0];
    for (int w = 1; w < 8; ++w) v = fmaxf(v, red[w]);
    red[8] = v;
  }
  __syncthreads();
  m = red[8];
  float s = 0.f;
  for (int i = tid; i < SEQ; i += 256) {
    float e = __expf(buf[i] - m);
    buf[i] = e;
    s += e;
  }
#pragma unroll
  for (int off = 16; off; off >>= 1) s += __shfl_xor(s, off, 32);
  __syncthreads();
  if ((tid & 31) == 0) red[tid >> 5] = s;
  __syncthreads();
  if (tid == 0) {
    float v = 0.f;
    for (int w = 0; w < 8; ++w) v += red[w];
    red[8] = v;
  }
  __syncthreads();
  float inv = 1.0f / red[8];
  for (int i = tid; i < SEQ; i += 256) row[i] = buf[i] * inv;
}

// ---------------- Kernel 4: attended = P @ V using Vt [B][H][D][S]
__global__ __launch_bounds__(128) void k_pv(const float* __restrict__ P,
                                            const __bf16* __restrict__ Vt,
                                            __bf16* __restrict__ attended) {
  const int bh = blockIdx.y;
  const int q0 = blockIdx.x * 128;
  const float*  Ph  = P + (size_t)bh * SEQ * SEQ;
  const __bf16* Vth = Vt + (size_t)bh * HD * SEQ;
  const int b = bh >> 4, h = bh & 15;
  __shared__ __bf16 Ps[128][40];       // [q][k]
  __shared__ __bf16 Vs[64][40];        // [d][k]  (rows of Vt, direct copy)
  const int tid = threadIdx.x, wave = tid >> 5, lane = tid & 31;
  v8f acc[8] = {};
  const int kend = q0 + 128;           // P is zero beyond the diagonal
  for (int kk0 = 0; kk0 < kend; kk0 += 32) {
    if (kk0 + 32 < kend)
      __builtin_prefetch(Ph + (size_t)(q0 + tid) * SEQ + kk0 + 32, 0, 3);
    __syncthreads();
    stage_async_bf16<64, 4>(&Vs[0][0], Vth + kk0, SEQ, 40);
    stage_a_f32<128>(&Ps[0][0], Ph + (size_t)q0 * SEQ + kk0, SEQ, 40);
    wait_async();
    __syncthreads();
    mma_2x4(acc, &Ps[0][0], wave * 32, 40, &Vs[0][0], 40);
  }
  const int kh = lane >> 4, nn = lane & 15;
#pragma unroll
  for (int h2 = 0; h2 < 2; ++h2) {
#pragma unroll
    for (int j = 0; j < 4; ++j) {
      const v8f& c = acc[h2 * 4 + j];
#pragma unroll
      for (int i = 0; i < 8; ++i) {
        int q = q0 + wave * 32 + h2 * 16 + i + 8 * kh;
        int d = j * 16 + nn;
        attended[(size_t)(b * SEQ + q) * EMB + h * HD + d] = (__bf16)c[i];
      }
    }
  }
}

// ---------------- Kernel 5: out = attended @ W_out + b_out (fp32 out)
__global__ __launch_bounds__(128) void k_out(const __bf16* __restrict__ attended,
                                             const float* __restrict__ Wout,
                                             const float* __restrict__ bout,
                                             float* __restrict__ out) {
  __shared__ __bf16 As[128][40];       // [M][K]
  __shared__ __bf16 Bs[64][40];        // [N][K]
  const int m0 = blockIdx.x * 128;
  const int n0 = blockIdx.y * 64;
  const int tid = threadIdx.x, wave = tid >> 5, lane = tid & 31;
  v8f acc[8] = {};
  for (int k0 = 0; k0 < EMB; k0 += 32) {
    if (k0 + 32 < EMB)
      __builtin_prefetch(Wout + (size_t)(k0 + 32 + (tid >> 2)) * EMB + n0, 0, 3);
    __syncthreads();
    stage_async_bf16<128, 4>(&As[0][0], attended + (size_t)m0 * EMB + k0, EMB, 40);
    stage_bT_f32(&Bs[0][0], Wout + (size_t)k0 * EMB + n0, EMB, 40);
    wait_async();
    __syncthreads();
    mma_2x4(acc, &As[0][0], wave * 32, 40, &Bs[0][0], 40);
  }
  const int kh = lane >> 4, nn = lane & 15;
#pragma unroll
  for (int h2 = 0; h2 < 2; ++h2) {
#pragma unroll
    for (int j = 0; j < 4; ++j) {
      int col = n0 + j * 16 + nn;
      float bias = bout[col];
      const v8f& c = acc[h2 * 4 + j];
#pragma unroll
      for (int i = 0; i < 8; ++i) {
        int row = m0 + wave * 32 + h2 * 16 + i + 8 * kh;
        out[(size_t)row * EMB + col] = c[i] + bias;
      }
    }
  }
}

extern "C" void kernel_launch(void* const* d_in, const int* in_sizes, int n_in,
                              void* d_out, int out_size, void* d_ws, size_t ws_size,
                              hipStream_t stream) {
  const float* x    = (const float*)d_in[0];
  const float* Wqkv = (const float*)d_in[1];
  const float* bqkv = (const float*)d_in[2];
  const float* Wout = (const float*)d_in[3];
  const float* bout = (const float*)d_in[4];

  float* out   = (float*)d_out;                       // (B,S,E) fp32
  float* attnW = out + (size_t)BATCH * SEQ * EMB;     // (B,H,S,S) fp32

  const size_t qkv_elems = (size_t)BATCH * HEADS * SEQ * HD;
  __bf16* Q        = (__bf16*)d_ws;                   // 8 MB
  __bf16* Kb       = Q  + qkv_elems;                  // 8 MB
  __bf16* Vt       = Kb + qkv_elems;                  // 8 MB (transposed [B][H][D][S])
  __bf16* attended = Vt + qkv_elems;                  // 8 MB (32 MB total ws)

  k_qkv    <<<dim3(32, 48),                 128, 0, stream>>>(x, Wqkv, bqkv, Q, Kb, Vt);
  k_scores <<<dim3(16, 32, BATCH * HEADS),  128, 0, stream>>>(Q, Kb, attnW);
  k_softmax<<<dim3(BATCH * HEADS * SEQ),    256, 0, stream>>>(attnW);
  k_pv     <<<dim3(16, BATCH * HEADS),      128, 0, stream>>>(attnW, Vt, attended);
  k_out    <<<dim3(32, 16),                 128, 0, stream>>>(attended, Wout, bout, out);
}
